// EfficientAttention_10960756540212
// MI455X (gfx1250) — matrix operand order repaired and checked by reading, not compiled, not run on hardware
//
#include <hip/hip_runtime.h>

// ---------------------------------------------------------------------------
// EfficientAttention for MI455X (gfx1250, wave32, WMMA)
// All GEMMs run as bf16 v_wmma_f32_16x16x32_bf16 with f32 accumulation.
// Every GEMM is expressed as C = A @ B^T with A (MxK) and B (NxK) row-major.
// K-loop is LDS-staged and double-buffered; global->LDS staging uses the
// CDNA5 async path (global_load_async_to_lds_b128 + s_wait_asynccnt).
// ---------------------------------------------------------------------------

typedef __bf16 bf16_t;
typedef __attribute__((ext_vector_type(16))) __bf16 v16bf;
typedef __attribute__((ext_vector_type(8)))  __bf16 v8bf;
typedef __attribute__((ext_vector_type(8)))  float  v8f;

#define DEVINL __device__ __forceinline__

// LDS tile: 128 rows x 32 halves of payload, padded to 40 halves (80B) per row
// to spread LDS banks across the 16-row fragment reads.
constexpr int LSTRIDE = 40;

// Async 16B copy global -> LDS (per active lane), tracked by ASYNCcnt.
// Generic LDS pointer truncated to 32 bits == LDS byte address (ISA 10.2).
DEVINL void async_copy_b128(void* lds_ptr, const void* gptr) {
  unsigned loff = (unsigned)(unsigned long long)lds_ptr;
  asm volatile("global_load_async_to_lds_b128 %0, %1, off"
               :
               : "v"(loff), "v"(gptr)
               : "memory");
}

DEVINL void wait_async0() {
#if __has_builtin(__builtin_amdgcn_s_wait_asynccnt)
  __builtin_amdgcn_s_wait_asynccnt(0);
#else
  asm volatile("s_wait_asynccnt 0x0" ::: "memory");
#endif
}

// A-fragment (16x32 bf16, MxK) from an LDS tile row block:
// lanes 0-15 row M=lane hold K {0..7,16..23},
// lanes 16-31 row M=lane-16 hold K {8..15,24..31}   (ISA 7.12.2)
DEVINL v16bf lds_frag_a(const bf16_t* base, int lane) {
  const bf16_t* p = base + (lane & 15) * LSTRIDE + ((lane >> 4) << 3);
  union { v16bf v; v8bf h[2]; } u;
  u.h[0] = *reinterpret_cast<const v8bf*>(p);
  u.h[1] = *reinterpret_cast<const v8bf*>(p + 16);
  return u.v;
}

// B-fragment (32x16 bf16, KxN) where column n of B == row n of row-major Bmat:
// lanes 0-15 col N=lane hold K {0..7,16..23}, lanes 16-31 hold K {8..15,24..31}.
DEVINL v16bf lds_frag_b(const bf16_t* base, int lane) {
  const bf16_t* p = base + (lane & 15) * LSTRIDE + ((lane >> 4) << 3);
  union { v16bf v; v8bf h[2]; } u;
  u.h[0] = *reinterpret_cast<const v8bf*>(p);
  u.h[1] = *reinterpret_cast<const v8bf*>(p + 16);
  return u.v;
}

// OUT_MODE: 0 = f32 row-major, 1 = bf16 row-major, 2 = bf16 transposed (C[n,m])
template <int OUT_MODE>
__global__ __launch_bounds__(256) void gemm_bt_kernel(
    const bf16_t* __restrict__ A, int lda, long long sA,
    const bf16_t* __restrict__ Bm, int ldb, long long sB,
    void* __restrict__ Cp, int ldc, long long sC, int K) {
  __shared__ bf16_t lsA[2][128 * LSTRIDE];
  __shared__ bf16_t lsB[2][128 * LSTRIDE];

  const int tid = threadIdx.x;
  const int lane = tid & 31;
  const int wave = tid >> 5;
  const int wm = (wave >> 2) * 64;  // wave M offset inside 128x128 block tile
  const int wn = (wave & 3) * 32;   // wave N offset
  const long long b = blockIdx.z;
  const int m0 = blockIdx.y * 128;
  const int n0 = blockIdx.x * 128;

  const bf16_t* Ag = A + b * sA + (long long)m0 * lda;
  const bf16_t* Bg = Bm + b * sB + (long long)n0 * ldb;

  // Cooperative tile stage: 128x32 halves = 512 x 16B chunks, 2 per thread.
  const int r0 = tid >> 2;            // rows 0..63
  const int r1 = 64 + (tid >> 2);     // rows 64..127
  const int oo = (tid & 3) * 8;       // 0,8,16,24 halves within the row

  v8f acc[4][2];
#pragma unroll
  for (int i = 0; i < 4; ++i)
#pragma unroll
    for (int j = 0; j < 2; ++j) acc[i][j] = (v8f)(0.0f);

  auto stage = [&](int q, int k) {
    async_copy_b128(&lsA[q][r0 * LSTRIDE + oo], Ag + (size_t)r0 * lda + k + oo);
    async_copy_b128(&lsA[q][r1 * LSTRIDE + oo], Ag + (size_t)r1 * lda + k + oo);
    async_copy_b128(&lsB[q][r0 * LSTRIDE + oo], Bg + (size_t)r0 * ldb + k + oo);
    async_copy_b128(&lsB[q][r1 * LSTRIDE + oo], Bg + (size_t)r1 * ldb + k + oo);
  };

  auto compute = [&](int p) {
    v16bf af[4], bfr[2];
#pragma unroll
    for (int i = 0; i < 4; ++i)
      af[i] = lds_frag_a(&lsA[p][(wm + i * 16) * LSTRIDE], lane);
#pragma unroll
    for (int j = 0; j < 2; ++j)
      bfr[j] = lds_frag_b(&lsB[p][(wn + j * 16) * LSTRIDE], lane);
#pragma unroll
    for (int i = 0; i < 4; ++i)
#pragma unroll
      for (int j = 0; j < 2; ++j)
        acc[i][j] = __builtin_amdgcn_wmma_f32_16x16x32_bf16(
            false, af[i], false, bfr[j], (short)0, acc[i][j], false, false);
  };

  // Prologue: async-stage k=0 into buffer 0.
  stage(0, 0);
  wait_async0();
  __syncthreads();

  int p = 0;
  for (int k0 = 0; k0 < K - 32; k0 += 32, p ^= 1) {
    stage(p ^ 1, k0 + 32);  // async DMA of next tile, overlapped with WMMAs
    compute(p);             // ds_load fragments from buffer p + 8 WMMAs
    wait_async0();          // this wave's DMA into buffer p^1 complete
    __syncthreads();        // all waves' DMA complete -> buffer p^1 ready
  }
  compute(p);  // last k-step

  // C/D layout: lane holds column (lane&15); rows (lane>>4)*8 + v, v=0..7
  const int col = lane & 15;
  const int rbase = (lane >> 4) * 8;
  const int mw = m0 + wm;
  const int nw = n0 + wn;

#pragma unroll
  for (int i = 0; i < 4; ++i) {
#pragma unroll
    for (int j = 0; j < 2; ++j) {
      if (OUT_MODE == 2) {
        // transposed bf16 store: 8 contiguous bf16 along m -> one 16B store
        bf16_t* C = (bf16_t*)Cp + b * sC;
        v8bf pack;
#pragma unroll
        for (int v = 0; v < 8; ++v) pack[v] = (bf16_t)acc[i][j][v];
        bf16_t* dst =
            C + (long long)(nw + j * 16 + col) * ldc + (mw + i * 16 + rbase);
        *reinterpret_cast<v8bf*>(dst) = pack;
      } else if (OUT_MODE == 1) {
        bf16_t* C = (bf16_t*)Cp + b * sC;
#pragma unroll
        for (int v = 0; v < 8; ++v)
          C[(long long)(mw + i * 16 + rbase + v) * ldc + (nw + j * 16 + col)] =
              (bf16_t)acc[i][j][v];
      } else {
        float* C = (float*)Cp + b * sC;
#pragma unroll
        for (int v = 0; v < 8; ++v)
          C[(long long)(mw + i * 16 + rbase + v) * ldc + (nw + j * 16 + col)] =
              acc[i][j][v];
      }
    }
  }
}

// ---------------------------------------------------------------------------
__global__ __launch_bounds__(256) void f32_to_bf16_kernel(
    const float* __restrict__ in, bf16_t* __restrict__ out, long long n) {
  long long i = (long long)blockIdx.x * blockDim.x + threadIdx.x;
  long long stride = (long long)gridDim.x * blockDim.x;
  for (; i < n; i += stride) out[i] = (bf16_t)in[i];
}

// In-place row softmax over last dim D==1024, one block (256 thr) per row.
__global__ __launch_bounds__(256) void softmax_row_kernel(bf16_t* data) {
  const int D = 1024;
  bf16_t* p = data + (long long)blockIdx.x * D;
  const int tid = threadIdx.x;
  __shared__ float red[256];

  float v[4];
  float m = -3.0e38f;
#pragma unroll
  for (int i = 0; i < 4; ++i) {
    v[i] = (float)p[tid + i * 256];
    m = fmaxf(m, v[i]);
  }
  red[tid] = m;
  __syncthreads();
  for (int s = 128; s > 0; s >>= 1) {
    if (tid < s) red[tid] = fmaxf(red[tid], red[tid + s]);
    __syncthreads();
  }
  m = red[0];
  __syncthreads();

  float sum = 0.0f;
#pragma unroll
  for (int i = 0; i < 4; ++i) {
    v[i] = __expf(v[i] - m);
    sum += v[i];
  }
  red[tid] = sum;
  __syncthreads();
  for (int s = 128; s > 0; s >>= 1) {
    if (tid < s) red[tid] += red[tid + s];
    __syncthreads();
  }
  const float inv = 1.0f / red[0];
#pragma unroll
  for (int i = 0; i < 4; ++i) p[tid + i * 256] = (bf16_t)(v[i] * inv);
}

// Column softmax (axis=N) of queries[b,n,d] with transposed output qT[b,d,n].
// One thread per (b,d); lane-consecutive d -> coalesced reads.
__global__ __launch_bounds__(256) void softmax_col_t_kernel(
    const bf16_t* __restrict__ q, bf16_t* __restrict__ qT, int Nn, int D) {
  const int d = blockIdx.x * 256 + threadIdx.x;
  const long long b = blockIdx.y;
  const bf16_t* src = q + b * (long long)Nn * D + d;
  bf16_t* dst = qT + b * (long long)Nn * D + (long long)d * Nn;

  float m = -3.0e38f;
  for (int n = 0; n < Nn; ++n) m = fmaxf(m, (float)src[(long long)n * D]);
  float s = 0.0f;
  for (int n = 0; n < Nn; ++n) s += __expf((float)src[(long long)n * D] - m);
  const float inv = 1.0f / s;
  for (int n = 0; n < Nn; ++n)
    dst[n] = (bf16_t)(__expf((float)src[(long long)n * D] - m) * inv);
}

// ---------------------------------------------------------------------------
extern "C" void kernel_launch(void* const* d_in, const int* in_sizes, int n_in,
                              void* d_out, int out_size, void* d_ws,
                              size_t ws_size, hipStream_t stream) {
  (void)in_sizes; (void)n_in; (void)out_size; (void)ws_size;
  const float* x  = (const float*)d_in[0];
  const float* Wk = (const float*)d_in[1];
  const float* Wq = (const float*)d_in[2];
  const float* Wv = (const float*)d_in[3];
  const float* Wr = (const float*)d_in[4];
  float* out = (float*)d_out;

  const long long Bb = 16, Nn = 1024, Dd = 1024;
  const long long BND = Bb * Nn * Dd;          // 16,777,216
  const long long MB = 1024ll * 1024ll;

  char* ws = (char*)d_ws;
  // Workspace layout (peak 168 MB, regions reused once dead):
  bf16_t* xb    = (bf16_t*)(ws + 0);           // x bf16          [0,32MB)
  bf16_t* Wkb   = (bf16_t*)(ws + 32 * MB);     // weights bf16    [32,40MB)
  bf16_t* Wqb   = Wkb + Dd * Dd;
  bf16_t* Wvb   = Wqb + Dd * Dd;
  bf16_t* Wrb   = Wvb + Dd * Dd;
  bf16_t* keysb = (bf16_t*)(ws + 40 * MB);     // keys -> key     [40,72MB)
  bf16_t* qb    = (bf16_t*)(ws + 72 * MB);     // queries         [72,104MB)
  bf16_t* valb  = (bf16_t*)(ws + 104 * MB);    // values          [104,136MB)
  bf16_t* qTb   = (bf16_t*)(ws + 136 * MB);    // query^T         [136,168MB)
  bf16_t* ctxTb = (bf16_t*)(ws + 0);           // reuse xb region after stage 1
  bf16_t* attb  = (bf16_t*)(ws + 72 * MB);     // reuse qb region after stage 3

  // Stage 0: f32 -> bf16 conversions
  f32_to_bf16_kernel<<<2048, 256, 0, stream>>>(x, xb, BND);
  f32_to_bf16_kernel<<<512, 256, 0, stream>>>(Wk, Wkb, Dd * Dd);
  f32_to_bf16_kernel<<<512, 256, 0, stream>>>(Wq, Wqb, Dd * Dd);
  f32_to_bf16_kernel<<<512, 256, 0, stream>>>(Wv, Wvb, Dd * Dd);
  f32_to_bf16_kernel<<<512, 256, 0, stream>>>(Wr, Wrb, Dd * Dd);

  // Stage 1: keys/queries/values = x @ W^T   (flattened M = B*N = 16384)
  {
    dim3 g((unsigned)(Dd / 128), (unsigned)(Bb * Nn / 128), 1);
    gemm_bt_kernel<1><<<g, 256, 0, stream>>>(xb, (int)Dd, 0, Wkb, (int)Dd, 0,
                                             keysb, (int)Dd, 0, (int)Dd);
    gemm_bt_kernel<1><<<g, 256, 0, stream>>>(xb, (int)Dd, 0, Wqb, (int)Dd, 0,
                                             qb, (int)Dd, 0, (int)Dd);
    gemm_bt_kernel<1><<<g, 256, 0, stream>>>(xb, (int)Dd, 0, Wvb, (int)Dd, 0,
                                             valb, (int)Dd, 0, (int)Dd);
  }

  // Stage 2: key = softmax(keys, axis=2), in place (row softmax)
  softmax_row_kernel<<<(unsigned)(Bb * Nn), 256, 0, stream>>>(keysb);

  // Stage 3: query = softmax(queries, axis=1), stored transposed qT[b,d,n]
  {
    dim3 g((unsigned)(Dd / 256), (unsigned)Bb, 1);
    softmax_col_t_kernel<<<g, 256, 0, stream>>>(qb, qTb, (int)Nn, (int)Dd);
  }

  // Stage 4: context[b] = key[b] @ value[b]^T, stored transposed ctxT[b,m,n]
  {
    dim3 g((unsigned)(Nn / 128), (unsigned)(Nn / 128), (unsigned)Bb);
    gemm_bt_kernel<2><<<g, 256, 0, stream>>>(
        keysb, (int)Dd, Nn * Dd, valb, (int)Dd, Nn * Dd,
        ctxTb, (int)Nn, Nn * Nn, (int)Dd);
  }

  // Stage 5: attended[b] = ctxT[b] @ qT[b]^T  -> attb[b,m,d] bf16
  {
    dim3 g((unsigned)(Dd / 128), (unsigned)(Nn / 128), (unsigned)Bb);
    gemm_bt_kernel<1><<<g, 256, 0, stream>>>(
        ctxTb, (int)Nn, Nn * Nn, qTb, (int)Nn, Dd * Nn,
        attb, (int)Dd, Nn * Dd, (int)Nn);
  }

  // Stage 6: out = attended @ Wr^T, f32 output (flattened M = B*N)
  {
    dim3 g((unsigned)(Dd / 128), (unsigned)(Bb * Nn / 128), 1);
    gemm_bt_kernel<0><<<g, 256, 0, stream>>>(attb, (int)Dd, 0, Wrb, (int)Dd, 0,
                                             out, (int)Dd, 0, (int)Dd);
  }
}
